// BidirectionalSpatialMambaBlock_83958020702531
// MI455X (gfx1250) — matrix operand order repaired and checked by reading, not compiled, not run on hardware
//
#include <hip/hip_runtime.h>
#include <math.h>

// ---------------- constants (from reference) ----------------
#define DIMD   512
#define STATE  16
#define BATCH  16
#define SEQL   1024          // 32*32
#define MROWS  (BATCH*SEQL)  // 16384
#define HIDF   1024          // ffn hidden

// ---------------- gfx1250 async Global->LDS path (guarded) ----------------
#ifndef HAVE_ASYNC_LDS
#if defined(__has_builtin)
#if __has_builtin(__builtin_amdgcn_global_load_async_to_lds_b128) && \
    __has_builtin(__builtin_amdgcn_s_wait_asynccnt)
#define HAVE_ASYNC_LDS 1
#endif
#endif
#endif
#ifndef HAVE_ASYNC_LDS
#define HAVE_ASYNC_LDS 0
#endif

#define AS1 __attribute__((address_space(1)))
#define AS3 __attribute__((address_space(3)))

// int4 vector type matching the async-LDS builtin's parameter type
typedef int v4i_t __attribute__((vector_size(16)));

// ---------------- bf16 vector types for WMMA ----------------
typedef __attribute__((ext_vector_type(16))) __bf16 v16bf;
typedef __attribute__((ext_vector_type(8)))  __bf16 v8bf;
typedef __attribute__((ext_vector_type(8)))  float  v8f;

static __device__ __forceinline__ unsigned short f2bf(float f) {
    union { float f; unsigned u; } v; v.f = f;
    unsigned r = v.u + 0x7fffu + ((v.u >> 16) & 1u);
    return (unsigned short)(r >> 16);
}

static __device__ __forceinline__ float gelu_exact(float x) {
    return 0.5f * x * (1.0f + erff(x * 0.70710678118654752f));
}
static __device__ __forceinline__ float softplus_f(float x) {
    return (x > 20.0f) ? x : log1pf(__expf(x));
}

static __device__ __forceinline__ v16bf combine8(const unsigned short* p0,
                                                 const unsigned short* p1) {
    v8bf lo = *(const v8bf*)p0;
    v8bf hi = *(const v8bf*)p1;
    v16bf r;
#pragma unroll
    for (int i = 0; i < 8; ++i) { r[i] = lo[i]; r[i + 8] = hi[i]; }
    return r;
}

// WMMA on one staged K-slab: 8 wmma per wave (4 M-tiles x 2 N-tiles)
static __device__ __forceinline__ void wmma_slab(
    const unsigned short* as, const unsigned short* ws,
    int waveM, int waveN, int laneLo, int klA, int kfB, v8f (*acc)[2])
{
    v16bf af[4], bfr[2];
#pragma unroll
    for (int mt = 0; mt < 4; ++mt) {
        int r = waveM * 64 + mt * 16 + laneLo;
        af[mt] = combine8(&as[r * 40 + klA], &as[r * 40 + klA + 16]);
    }
#pragma unroll
    for (int nt = 0; nt < 2; ++nt) {
        int n = waveN * 32 + nt * 16 + laneLo;
        bfr[nt] = combine8(&ws[n * 40 + kfB], &ws[n * 40 + kfB + 16]);
    }
#pragma unroll
    for (int mt = 0; mt < 4; ++mt)
#pragma unroll
        for (int nt = 0; nt < 2; ++nt)
            acc[mt][nt] = __builtin_amdgcn_wmma_f32_16x16x32_bf16(
                false, af[mt], false, bfr[nt], (short)0, acc[mt][nt], false, false);
}

// ============================================================
// WMMA bf16 GEMM: C[M,N] = act(A[M,K] * Wt[N,K]^T + bias) (+res)
// A row-major bf16 (lda halves), Wt row-major (N rows of K) bf16.
// Optional fp32 output Cf and/or bf16 output Ch (with column offset).
// Block tile 128x128, 8 waves, each wave 64x32 (4x2 16x16 tiles).
// Async path: double-buffered LDS, DMA of slab k+1 overlaps WMMA of slab k.
// ============================================================
template <int ACT>  // 0=none 1=gelu 2=softplus
__global__ __launch_bounds__(256) void gemm_bf16_k(
    const unsigned short* __restrict__ A, int lda,
    const unsigned short* __restrict__ Wt, int ldw,
    const float* __restrict__ bias,
    const float* __restrict__ res, int ldres,
    float* __restrict__ Cf, int ldcf,
    unsigned short* __restrict__ Ch, int ldch, int chOff,
    int K)
{
    const int tid   = threadIdx.x;
    const int lane  = tid & 31;
    const int wave  = tid >> 5;
    const int waveM = wave >> 2;   // 0..1
    const int waveN = wave & 3;    // 0..3
    const int rowBase = blockIdx.y * 128;
    const int colBase = blockIdx.x * 128;

    v8f acc[4][2];
#pragma unroll
    for (int mt = 0; mt < 4; ++mt)
#pragma unroll
        for (int nt = 0; nt < 2; ++nt)
#pragma unroll
            for (int i = 0; i < 8; ++i) acc[mt][nt][i] = 0.0f;

    const int lr = tid >> 1;          // staging row 0..127
    const int lc = (tid & 1) * 16;    // 0 or 16 (halves)

    const int laneLo = lane & 15;
    const int klA = (lane >> 4) * 8;   // A frag K base (0 or 8)
    const int kfB = (lane >> 4) * 8;   // B frag K base (0 or 8)

#if HAVE_ASYNC_LDS
    __shared__ __align__(16) unsigned short As[2][128 * 40];
    __shared__ __align__(16) unsigned short Ws[2][128 * 40];

    auto stage = [&](int k0, int buf) {
        // each thread DMAs 32B of A-tile and 32B of W-tile straight into LDS
        AS1 v4i_t* ga = (AS1 v4i_t*)(A  + (size_t)(rowBase + lr) * lda + k0 + lc);
        AS1 v4i_t* gw = (AS1 v4i_t*)(Wt + (size_t)(colBase + lr) * ldw + k0 + lc);
        AS3 v4i_t* la = (AS3 v4i_t*)&As[buf][lr * 40 + lc];
        AS3 v4i_t* lw = (AS3 v4i_t*)&Ws[buf][lr * 40 + lc];
        __builtin_amdgcn_global_load_async_to_lds_b128(ga, la, 0,  0);
        __builtin_amdgcn_global_load_async_to_lds_b128(ga, la, 16, 0);
        __builtin_amdgcn_global_load_async_to_lds_b128(gw, lw, 0,  0);
        __builtin_amdgcn_global_load_async_to_lds_b128(gw, lw, 16, 0);
    };

    const int nk = K >> 5;
    stage(0, 0);
    __builtin_amdgcn_s_wait_asynccnt(0);
    __syncthreads();
    for (int kk = 0; kk < nk; ++kk) {
        const int buf = kk & 1;
        if (kk + 1 < nk) stage((kk + 1) << 5, buf ^ 1);   // DMA next slab
        wmma_slab(&As[buf][0], &Ws[buf][0], waveM, waveN, laneLo, klA, kfB, acc);
        if (kk + 1 < nk) {
            __builtin_amdgcn_s_wait_asynccnt(0);          // my wave's DMA landed
            __syncthreads();                              // everyone's landed / RAW+WAR guard
        }
    }
#else
    __shared__ __align__(16) unsigned short As1[128 * 40];
    __shared__ __align__(16) unsigned short Ws1[128 * 40];

    for (int k0 = 0; k0 < K; k0 += 32) {
        const uint4* ga = (const uint4*)(A  + (size_t)(rowBase + lr) * lda + k0 + lc);
        const uint4* gw = (const uint4*)(Wt + (size_t)(colBase + lr) * ldw + k0 + lc);
        uint4 a0 = ga[0], a1 = ga[1];
        uint4 w0 = gw[0], w1 = gw[1];
        if (k0 + 32 < K) {
            __builtin_prefetch(A  + (size_t)(rowBase + lr) * lda + k0 + 32 + lc, 0, 0);
            __builtin_prefetch(Wt + (size_t)(colBase + lr) * ldw + k0 + 32 + lc, 0, 0);
        }
        __syncthreads();
        *(uint4*)&As1[lr * 40 + lc]     = a0;
        *(uint4*)&As1[lr * 40 + lc + 8] = a1;
        *(uint4*)&Ws1[lr * 40 + lc]     = w0;
        *(uint4*)&Ws1[lr * 40 + lc + 8] = w1;
        __syncthreads();
        wmma_slab(As1, Ws1, waveM, waveN, laneLo, klA, kfB, acc);
    }
#endif

    const int laneHi = lane >> 4;
#pragma unroll
    for (int mt = 0; mt < 4; ++mt) {
#pragma unroll
        for (int nt = 0; nt < 2; ++nt) {
            int gn = colBase + waveN * 32 + nt * 16 + laneLo;
            float bv = bias ? bias[gn] : 0.0f;
#pragma unroll
            for (int i = 0; i < 8; ++i) {
                int gm = rowBase + waveM * 64 + mt * 16 + laneHi * 8 + i;
                float v = acc[mt][nt][i] + bv;
                if (ACT == 1) v = gelu_exact(v);
                else if (ACT == 2) v = softplus_f(v);
                if (res) v += res[(size_t)gm * ldres + gn];
                if (Cf)  Cf[(size_t)gm * ldcf + gn] = v;
                if (Ch)  Ch[(size_t)gm * ldch + chOff + gn] = f2bf(v);
            }
        }
    }
}

// ---------------- small utility kernels ----------------
__global__ void fill_u16_k(unsigned short* p, int n, unsigned short v) {
    int i = blockIdx.x * 256 + threadIdx.x;
    if (i < n) p[i] = v;
}

// Wt[(rowOff+n)*ldw + colOff + k] = bf16(W[k*N + n])   (W is K x N fp32)
__global__ void transpose_bf16_k(const float* __restrict__ W, int K, int N,
                                 unsigned short* __restrict__ Wt, int ldw,
                                 int rowOff, int colOff) {
    int i = blockIdx.x * 256 + threadIdx.x;
    if (i >= K * N) return;
    int k = i / N, n = i % N;
    Wt[(size_t)(rowOff + n) * ldw + colOff + k] = f2bf(W[(size_t)k * N + n]);
}

__global__ void pack_bias_bc_k(const float* bBf, const float* bCf,
                               const float* bBb, const float* bCb, float* out) {
    int i = threadIdx.x;  // 128 threads
    float v = 0.0f;
    if      (i < 16) v = bBf[i];
    else if (i < 32) v = bCf[i - 16];
    else if (i < 48) v = bBb[i - 32];
    else if (i < 64) v = bCb[i - 48];
    out[i] = v;
}

// pos MLP stage 1: c1[l][j] = bf16(gelu(pos(l) . W_p1[:,j] + b_p1[j]))
__global__ void pos_mlp1_k(const float* __restrict__ W_p1, const float* __restrict__ b_p1,
                           const int* hp, const int* wp, unsigned short* __restrict__ c1) {
    int l = blockIdx.x, j = threadIdx.x;  // 1024 blocks x 512 threads
    int h = hp[0], w = wp[0];
    int r = l / w, c = l % w;
    float yy = -1.0f + 2.0f * (float)r / (float)(h - 1);
    float xx = -1.0f + 2.0f * (float)c / (float)(w - 1);
    float p[6] = { yy, xx, yy * xx, yy * yy, xx * xx,
                   sqrtf(xx * xx + yy * yy + 1e-6f) };
    float acc = b_p1[j];
#pragma unroll
    for (int i = 0; i < 6; ++i) acc += p[i] * W_p1[i * DIMD + j];
    c1[(size_t)l * DIMD + j] = f2bf(gelu_exact(acc));
}

static __device__ __forceinline__ float block_sum(float v, float* red, int tid) {
    red[tid] = v; __syncthreads();
    for (int s = 128; s > 0; s >>= 1) {
        if (tid < s) red[tid] += red[tid + s];
        __syncthreads();
    }
    float r = red[0]; __syncthreads();
    return r;
}

// E1: scan_in = LN(tokens)*g+b + cond ; fill Acat = [scan_in|cond] bf16,
//     Mixcat[:,1024:1536] = cond bf16
__global__ __launch_bounds__(256) void ln_cond_k(
    const float* __restrict__ tokens, const float* __restrict__ g,
    const float* __restrict__ bta, const float* __restrict__ cond,
    float* __restrict__ scan_in, unsigned short* __restrict__ Acat,
    unsigned short* __restrict__ Mixcat)
{
    __shared__ float red[256];
    int m = blockIdx.x, tid = threadIdx.x;
    int l = m & (SEQL - 1);
    float v0 = tokens[(size_t)m * DIMD + tid];
    float v1 = tokens[(size_t)m * DIMD + tid + 256];
    float mean = block_sum(v0 + v1, red, tid) * (1.0f / DIMD);
    float d0 = v0 - mean, d1 = v1 - mean;
    float var = block_sum(d0 * d0 + d1 * d1, red, tid) * (1.0f / DIMD);
    float rstd = rsqrtf(var + 1e-5f);
#pragma unroll
    for (int q = 0; q < 2; ++q) {
        int j = tid + q * 256;
        float x = (q == 0) ? v0 : v1;
        float xn = (x - mean) * rstd * g[j] + bta[j];
        float cv = cond[(size_t)l * DIMD + j];
        float si = xn + cv;
        scan_in[(size_t)m * DIMD + j] = si;
        Acat[(size_t)m * (2 * DIMD) + j]        = f2bf(si);
        Acat[(size_t)m * (2 * DIMD) + DIMD + j] = f2bf(cv);
        Mixcat[(size_t)m * (3 * DIMD) + 2 * DIMD + j] = f2bf(cv);
    }
}

// E2: t = LN(tokens+mixed)*out_g+out_b (fp32) ; tn = LN(t)*ff_g+ff_b (bf16)
__global__ __launch_bounds__(256) void ln2_k(
    const float* __restrict__ tokens, const float* __restrict__ mixed,
    const float* __restrict__ og, const float* __restrict__ ob,
    const float* __restrict__ fg, const float* __restrict__ fb,
    float* __restrict__ tbuf, unsigned short* __restrict__ tn)
{
    __shared__ float red[256];
    int m = blockIdx.x, tid = threadIdx.x;
    float u0 = tokens[(size_t)m * DIMD + tid]       + mixed[(size_t)m * DIMD + tid];
    float u1 = tokens[(size_t)m * DIMD + tid + 256] + mixed[(size_t)m * DIMD + tid + 256];
    float mean = block_sum(u0 + u1, red, tid) * (1.0f / DIMD);
    float d0 = u0 - mean, d1 = u1 - mean;
    float var = block_sum(d0 * d0 + d1 * d1, red, tid) * (1.0f / DIMD);
    float rstd = rsqrtf(var + 1e-5f);
    float t0 = d0 * rstd * og[tid]       + ob[tid];
    float t1 = d1 * rstd * og[tid + 256] + ob[tid + 256];
    tbuf[(size_t)m * DIMD + tid]       = t0;
    tbuf[(size_t)m * DIMD + tid + 256] = t1;
    float mean2 = block_sum(t0 + t1, red, tid) * (1.0f / DIMD);
    float e0 = t0 - mean2, e1 = t1 - mean2;
    float var2 = block_sum(e0 * e0 + e1 * e1, red, tid) * (1.0f / DIMD);
    float rstd2 = rsqrtf(var2 + 1e-5f);
    tn[(size_t)m * DIMD + tid]       = f2bf(e0 * rstd2 * fg[tid]       + fb[tid]);
    tn[(size_t)m * DIMD + tid + 256] = f2bf(e1 * rstd2 * fg[tid + 256] + fb[tid + 256]);
}

// Selective scan: one thread per (b,d) channel, 1024 sequential steps.
// B/C rows staged in LDS in 64-token chunks. Writes bf16 into Mixcat slice.
__global__ __launch_bounds__(256) void scan_k(
    const float* __restrict__ scan_in, const float* __restrict__ dt,
    const float* __restrict__ BC, int bcCol,
    const float* __restrict__ A_log, const float* __restrict__ Dp,
    unsigned short* __restrict__ mixOut, int mixColOff, int bwd)
{
    __shared__ float bcs[64 * 32];   // 64 tokens x (16 B + 16 C)
    const int tid = threadIdx.x;
    const int b = blockIdx.x >> 1;
    const int d = ((blockIdx.x & 1) << 8) + tid;

    float a[STATE], h[STATE];
#pragma unroll
    for (int s = 0; s < STATE; ++s) {
        a[s] = -__expf(A_log[(size_t)d * STATE + s]);
        h[s] = 0.0f;
    }
    const float Dv = Dp[d];

    for (int c = 0; c < SEQL / 64; ++c) {
        __syncthreads();
        for (int idx = tid; idx < 64 * 32; idx += 256) {
            int slot = idx >> 5, col = idx & 31;
            int to = c * 64 + slot;
            int tok = bwd ? (SEQL - 1 - to) : to;
            bcs[idx] = BC[((size_t)b * SEQL + tok) * 128 + bcCol + col];
        }
        __syncthreads();
        for (int i = 0; i < 64; ++i) {
            int to = c * 64 + i;
            int tok = bwd ? (SEQL - 1 - to) : to;
            size_t m = (size_t)b * SEQL + tok;
            float x   = scan_in[m * DIMD + d];
            float dtv = dt[m * DIMD + d];
            float dx  = dtv * x;
            const float* Bc = &bcs[i * 32];
            const float* Cc = Bc + 16;
            float y = 0.0f;
#pragma unroll
            for (int s = 0; s < STATE; ++s) {
                h[s] = __expf(dtv * a[s]) * h[s] + dx * Bc[s];
                y += h[s] * Cc[s];
            }
            mixOut[m * (3 * DIMD) + mixColOff + d] = f2bf(y + x * Dv);
        }
    }
}

// ============================================================
extern "C" void kernel_launch(void* const* d_in, const int* in_sizes, int n_in,
                              void* d_out, int out_size, void* d_ws, size_t ws_size,
                              hipStream_t stream) {
    (void)in_sizes; (void)n_in; (void)out_size; (void)ws_size;
    // ---- input pointers (setup_inputs dict order, nested dicts flattened) ----
    const float* tokens = (const float*)d_in[0];
    const float* in_g   = (const float*)d_in[1];
    const float* in_b   = (const float*)d_in[2];
    const float* W_p1   = (const float*)d_in[3];
    const float* b_p1   = (const float*)d_in[4];
    const float* W_p2   = (const float*)d_in[5];
    const float* b_p2   = (const float*)d_in[6];
    // fwd cell
    const float* A_log_f = (const float*)d_in[7];
    const float* D_f     = (const float*)d_in[8];
    const float* W_dt_f  = (const float*)d_in[9];
    const float* W_cdt_f = (const float*)d_in[10];
    const float* b_dt_f  = (const float*)d_in[11];
    const float* W_B_f   = (const float*)d_in[12];
    const float* b_B_f   = (const float*)d_in[13];
    const float* W_C_f   = (const float*)d_in[14];
    const float* b_C_f   = (const float*)d_in[15];
    // bwd cell
    const float* A_log_b = (const float*)d_in[16];
    const float* D_b     = (const float*)d_in[17];
    const float* W_dt_b  = (const float*)d_in[18];
    const float* W_cdt_b = (const float*)d_in[19];
    const float* b_dt_b  = (const float*)d_in[20];
    const float* W_B_b   = (const float*)d_in[21];
    const float* b_B_b   = (const float*)d_in[22];
    const float* W_C_b   = (const float*)d_in[23];
    const float* b_C_b   = (const float*)d_in[24];
    const float* W_mix  = (const float*)d_in[25];
    const float* b_mix  = (const float*)d_in[26];
    const float* out_g  = (const float*)d_in[27];
    const float* out_b  = (const float*)d_in[28];
    const float* ff_g   = (const float*)d_in[29];
    const float* ff_b   = (const float*)d_in[30];
    const float* W_f1   = (const float*)d_in[31];
    const float* b_f1   = (const float*)d_in[32];
    const float* W_f2   = (const float*)d_in[33];
    const float* b_f2   = (const float*)d_in[34];
    const int*   hptr   = (const int*)d_in[35];
    const int*   wptr   = (const int*)d_in[36];
    float* outp = (float*)d_out;

    // ---- workspace carve-up (256B aligned) ----
    char* wsb = (char*)d_ws;
    size_t off = 0;
    auto alloc = [&](size_t bytes) -> void* {
        off = (off + 255) & ~(size_t)255;
        void* p = wsb + off;
        off += bytes;
        return p;
    };
    unsigned short* c1      = (unsigned short*)alloc((size_t)SEQL * DIMD * 2);
    float*          cond    = (float*)alloc((size_t)SEQL * DIMD * 4);
    unsigned short* Wt_p2   = (unsigned short*)alloc((size_t)DIMD * DIMD * 2);
    unsigned short* Wt_dt_f = (unsigned short*)alloc((size_t)DIMD * 2 * DIMD * 2);
    unsigned short* Wt_dt_b = (unsigned short*)alloc((size_t)DIMD * 2 * DIMD * 2);
    unsigned short* Wt_bc   = (unsigned short*)alloc((size_t)128 * DIMD * 2);
    float*          bias_bc = (float*)alloc(128 * 4);
    unsigned short* Wt_mix  = (unsigned short*)alloc((size_t)DIMD * 3 * DIMD * 2);
    unsigned short* Wt_f1   = (unsigned short*)alloc((size_t)HIDF * DIMD * 2);
    unsigned short* Wt_f2   = (unsigned short*)alloc((size_t)DIMD * HIDF * 2);
    float*          scan_in = (float*)alloc((size_t)MROWS * DIMD * 4);
    unsigned short* Acat    = (unsigned short*)alloc((size_t)MROWS * 2 * DIMD * 2);
    float*          dt_f    = (float*)alloc((size_t)MROWS * DIMD * 4);
    float*          dt_b    = (float*)alloc((size_t)MROWS * DIMD * 4);
    float*          BC      = (float*)alloc((size_t)MROWS * 128 * 4);
    unsigned short* Mixcat  = (unsigned short*)alloc((size_t)MROWS * 3 * DIMD * 2);
    float*          mixed   = (float*)alloc((size_t)MROWS * DIMD * 4);
    float*          tbuf    = (float*)alloc((size_t)MROWS * DIMD * 4);
    unsigned short* tn      = (unsigned short*)alloc((size_t)MROWS * DIMD * 2);
    unsigned short* ffh     = (unsigned short*)alloc((size_t)MROWS * HIDF * 2);

    // ---- weight prep: transpose fp32 -> bf16 (N-major, contiguous K) ----
    fill_u16_k<<<(128 * DIMD + 255) / 256, 256, 0, stream>>>(Wt_bc, 128 * DIMD, 0);
    auto tr = [&](const float* W, int K, int N, unsigned short* Wt, int ldw,
                  int rowOff, int colOff) {
        transpose_bf16_k<<<(K * N + 255) / 256, 256, 0, stream>>>(W, K, N, Wt, ldw, rowOff, colOff);
    };
    tr(W_p2,   DIMD, DIMD, Wt_p2,   DIMD,     0, 0);
    tr(W_dt_f, DIMD, DIMD, Wt_dt_f, 2 * DIMD, 0, 0);
    tr(W_cdt_f,DIMD, DIMD, Wt_dt_f, 2 * DIMD, 0, DIMD);
    tr(W_dt_b, DIMD, DIMD, Wt_dt_b, 2 * DIMD, 0, 0);
    tr(W_cdt_b,DIMD, DIMD, Wt_dt_b, 2 * DIMD, 0, DIMD);
    tr(W_B_f,  DIMD, STATE, Wt_bc,  DIMD,  0, 0);
    tr(W_C_f,  DIMD, STATE, Wt_bc,  DIMD, 16, 0);
    tr(W_B_b,  DIMD, STATE, Wt_bc,  DIMD, 32, 0);
    tr(W_C_b,  DIMD, STATE, Wt_bc,  DIMD, 48, 0);
    tr(W_mix,  3 * DIMD, DIMD, Wt_mix, 3 * DIMD, 0, 0);
    tr(W_f1,   DIMD, HIDF, Wt_f1,  DIMD, 0, 0);
    tr(W_f2,   HIDF, DIMD, Wt_f2,  HIDF, 0, 0);
    pack_bias_bc_k<<<1, 128, 0, stream>>>(b_B_f, b_C_f, b_B_b, b_C_b, bias_bc);

    // ---- cond MLP ----
    pos_mlp1_k<<<SEQL, DIMD, 0, stream>>>(W_p1, b_p1, hptr, wptr, c1);
    // cond = c1 @ W_p2 + b_p2   (M=1024, N=512, K=512)
    gemm_bf16_k<0><<<dim3(DIMD / 128, SEQL / 128), 256, 0, stream>>>(
        c1, DIMD, Wt_p2, DIMD, b_p2, nullptr, 0, cond, DIMD, nullptr, 0, 0, DIMD);

    // ---- LN + cond add; build Acat / Mixcat cond slice ----
    ln_cond_k<<<MROWS, 256, 0, stream>>>(tokens, in_g, in_b, cond, scan_in, Acat, Mixcat);

    // ---- dt projections: softplus([scan_in|cond] @ [W_dt;W_cdt] + b_dt) ----
    gemm_bf16_k<2><<<dim3(DIMD / 128, MROWS / 128), 256, 0, stream>>>(
        Acat, 2 * DIMD, Wt_dt_f, 2 * DIMD, b_dt_f, nullptr, 0, dt_f, DIMD,
        nullptr, 0, 0, 2 * DIMD);
    gemm_bf16_k<2><<<dim3(DIMD / 128, MROWS / 128), 256, 0, stream>>>(
        Acat, 2 * DIMD, Wt_dt_b, 2 * DIMD, b_dt_b, nullptr, 0, dt_b, DIMD,
        nullptr, 0, 0, 2 * DIMD);

    // ---- B/C projections (all four packed; N padded to 128) ----
    gemm_bf16_k<0><<<dim3(1, MROWS / 128), 256, 0, stream>>>(
        Acat, 2 * DIMD, Wt_bc, DIMD, bias_bc, nullptr, 0, BC, 128,
        nullptr, 0, 0, DIMD);

    // ---- selective scans (fwd + bwd), write bf16 into Mixcat ----
    scan_k<<<2 * BATCH, 256, 0, stream>>>(scan_in, dt_f, BC, 0,  A_log_f, D_f,
                                          Mixcat, 0,    0);
    scan_k<<<2 * BATCH, 256, 0, stream>>>(scan_in, dt_b, BC, 32, A_log_b, D_b,
                                          Mixcat, DIMD, 1);

    // ---- mix: [fwd|bwd|cond] @ W_mix + b_mix ----
    gemm_bf16_k<0><<<dim3(DIMD / 128, MROWS / 128), 256, 0, stream>>>(
        Mixcat, 3 * DIMD, Wt_mix, 3 * DIMD, b_mix, nullptr, 0, mixed, DIMD,
        nullptr, 0, 0, 3 * DIMD);

    // ---- two layer norms -> tbuf (fp32), tn (bf16) ----
    ln2_k<<<MROWS, 256, 0, stream>>>(tokens, mixed, out_g, out_b, ff_g, ff_b, tbuf, tn);

    // ---- FFN: gelu(tn @ W_f1 + b_f1) ----
    gemm_bf16_k<1><<<dim3(HIDF / 128, MROWS / 128), 256, 0, stream>>>(
        tn, DIMD, Wt_f1, DIMD, b_f1, nullptr, 0, nullptr, 0, ffh, HIDF, 0, DIMD);

    // ---- out = tbuf + ffh @ W_f2 + b_f2 ----
    gemm_bf16_k<0><<<dim3(DIMD / 128, MROWS / 128), 256, 0, stream>>>(
        ffh, HIDF, Wt_f2, HIDF, b_f2, tbuf, DIMD, outp, DIMD, nullptr, 0, 0, HIDF);
}
